// BaseLayer_16063177687659
// MI455X (gfx1250) — matrix operand order repaired and checked by reading, not compiled, not run on hardware
//
#include <hip/hip_runtime.h>
#include <cstdint>

typedef unsigned short u16;
typedef __bf16 bfrag __attribute__((ext_vector_type(16)));
typedef float  ffrag __attribute__((ext_vector_type(8)));

#define LTOK   8192
#define DMODEL 512
#define HFFN   2048
#define OSTR   516   // padded O-accumulator row stride (dwords) to avoid LDS bank conflicts

// ---------- helpers ----------
__device__ __forceinline__ u16 f2bf(float f) {
  unsigned u = __float_as_uint(f);
  unsigned r = (u + 0x7fffu + ((u >> 16) & 1u)) >> 16;  // round-to-nearest-even
  return (u16)r;
}

__device__ __forceinline__ ffrag wmma_bf16(bfrag a, bfrag b, ffrag c) {
  // D = A(16x32 bf16) * B(32x16 bf16) + C(16x16 f32)
  return __builtin_amdgcn_wmma_f32_16x16x32_bf16(false, a, false, b, (short)0, c, false, false);
}

union FragU { int4 q[2]; bfrag f; };

// A-fragment (16x32 bf16) from row-major bf16 matrix.
// Lane layout: row = lane&15; K = 0..15 (lane<16) / 16..31 (lane>=16), contiguous
__device__ __forceinline__ bfrag load_afrag(const u16* base, int row0, int ld, int k0, int lane) {
  int m = lane & 15, hi = lane >> 4;
  const u16* p = base + (size_t)(row0 + m) * ld + k0 + hi * 16;
  FragU u;
  u.q[0] = *reinterpret_cast<const int4*>(p);
  u.q[1] = *reinterpret_cast<const int4*>(p + 8);
  return u.f;
}

// B-fragment (32x16 bf16) from B^T stored row-major (N x K) bf16.
// Lane layout: col n = lane&15; K = 0..15 (lane<16) / 16..31 (lane>=16), contiguous in B^T row.
__device__ __forceinline__ bfrag load_bfrag(const u16* base, int n0, int ld, int k0, int lane) {
  int nn = lane & 15, hi = lane >> 4;
  const u16* p = base + (size_t)(n0 + nn) * ld + k0 + hi * 16;
  FragU u;
  u.q[0] = *reinterpret_cast<const int4*>(p);
  u.q[1] = *reinterpret_cast<const int4*>(p + 8);
  return u.f;
}

// B-fragment for V = bf16 xcat (row-major L x D): column-gather of 16-bit elements.
__device__ __forceinline__ bfrag load_vfrag(const u16* xv, int j0, int n0, int lane) {
  int nn = lane & 15, hi = lane >> 4;
  const u16* col = xv + n0 + nn;
  int kb = j0 + hi * 16;
  union { unsigned d[8]; bfrag f; } u;
#pragma unroll
  for (int t = 0; t < 8; ++t) {
    unsigned lo = col[(size_t)(kb + 2 * t) * DMODEL];
    unsigned hw = col[(size_t)(kb + 2 * t + 1) * DMODEL];
    u.d[t] = lo | (hw << 16);
  }
  return u.f;
}

// A-fragment of P-tilde (16x32 bf16) from per-wave LDS tile Ps[16][32]
__device__ __forceinline__ bfrag load_pfrag(const u16* Ps, int lane) {
  int m = lane & 15, hi = lane >> 4;
  const u16* p = Ps + m * 32 + hi * 16;
  FragU u;
  u.q[0] = *reinterpret_cast<const int4*>(p);
  u.q[1] = *reinterpret_cast<const int4*>(p + 8);
  return u.f;
}

// ---------- utility kernels ----------
// column (per-feature) mean / rstd over all tokens
__global__ __launch_bounds__(256) void colstats_kernel(const float* __restrict__ x,
                                                       float* mu, float* rsig, int Lr, int Dc) {
  __shared__ float s1[256];
  __shared__ float s2[256];
  int d = blockIdx.x, t = threadIdx.x;
  float a = 0.f, b = 0.f;
  for (int i = t; i < Lr; i += 256) {
    float v = x[(size_t)i * Dc + d];
    a += v; b += v * v;
  }
  s1[t] = a; s2[t] = b;
  __syncthreads();
  for (int s = 128; s > 0; s >>= 1) {
    if (t < s) { s1[t] += s1[t + s]; s2[t] += s2[t + s]; }
    __syncthreads();
  }
  if (t == 0) {
    float m = s1[0] / (float)Lr;
    float var = s2[0] / (float)Lr - m * m;
    mu[d] = m;
    rsig[d] = rsqrtf(var + 1e-6f);
  }
}

__global__ __launch_bounds__(256) void ln_bf16_kernel(const float* __restrict__ x,
                                                      const float* __restrict__ mu,
                                                      const float* __restrict__ rsig,
                                                      u16* __restrict__ out, int total) {
  int i = blockIdx.x * 256 + threadIdx.x;
  if (i >= total) return;
  int d = i & (DMODEL - 1);
  out[i] = f2bf((x[i] - mu[d]) * rsig[d]);
}

__global__ __launch_bounds__(256) void cvt_bf16_kernel(const float* __restrict__ x,
                                                       u16* __restrict__ out, int total) {
  int i = blockIdx.x * 256 + threadIdx.x;
  if (i >= total) return;
  out[i] = f2bf(x[i]);
}

// dst (N x K, row-major bf16) = transpose of src[k][c0+n] (f32, leading dim src_ld)
__global__ __launch_bounds__(256) void transpose_bf16_kernel(const float* __restrict__ src, int src_ld,
                                                             int c0, int Kr, int Nc,
                                                             u16* __restrict__ dst) {
  int idx = blockIdx.x * 256 + threadIdx.x;
  if (idx >= Kr * Nc) return;
  int n = idx / Kr, k = idx - n * Kr;
  dst[idx] = f2bf(src[(size_t)k * src_ld + c0 + n]);
}

__global__ void seg_init_kernel(int* seg) {
  if (threadIdx.x < 16) seg[threadIdx.x] = 0x7fffffff;
}

__global__ __launch_bounds__(256) void seg_scan_kernel(const int* __restrict__ sep, int* seg, int Lr) {
  int i = blockIdx.x * 256 + threadIdx.x;
  if (i >= Lr) return;
  int s = sep[i];
  if (i == 0 || sep[i - 1] != s) atomicMin(&seg[s], i);
}

// ---------- generic WMMA GEMM: C = act(A @ Bt^T + bias) (+res) ----------
// A: (M x K) bf16 row-major. Bt: (N x K) bf16 row-major (i.e. B transposed).
// Each wave owns a 32x64 output tile: 8 wmma per k-step vs 12 b128 loads
// (2 A-frags + 4 B-frags, each reused) for better matrix-pipe issue density.
template <bool RELU, bool OUTBF16, bool RES>
__global__ __launch_bounds__(256) void gemm_bf16(const u16* __restrict__ A, const u16* __restrict__ Bt,
                                                 const float* __restrict__ bias,
                                                 const float* __restrict__ res,
                                                 void* __restrict__ Cout, int M, int N, int K) {
  int lane = threadIdx.x & 31;
  int wave = threadIdx.x >> 5;
  long tile = (long)blockIdx.x * 8 + wave;
  int nt4 = N >> 6;
  long T = (long)(M >> 5) * nt4;
  if (tile >= T) return;
  int mt = (int)(tile / nt4);
  int n4 = (int)(tile % nt4);
  int m0 = mt << 5, n0 = n4 << 6;

  ffrag acc[2][4];
#pragma unroll
  for (int mi = 0; mi < 2; ++mi)
#pragma unroll
    for (int t = 0; t < 4; ++t)
#pragma unroll
      for (int r = 0; r < 8; ++r) acc[mi][t][r] = 0.f;

  for (int k0 = 0; k0 < K; k0 += 32) {
    bfrag a0 = load_afrag(A, m0, K, k0, lane);
    bfrag a1 = load_afrag(A, m0 + 16, K, k0, lane);
#pragma unroll
    for (int t = 0; t < 4; ++t) {
      bfrag bf = load_bfrag(Bt, n0 + t * 16, K, k0, lane);
      acc[0][t] = wmma_bf16(a0, bf, acc[0][t]);
      acc[1][t] = wmma_bf16(a1, bf, acc[1][t]);
    }
  }

  int nn = lane & 15, hi = lane >> 4;
#pragma unroll
  for (int mi = 0; mi < 2; ++mi) {
#pragma unroll
    for (int t = 0; t < 4; ++t) {
      int n = n0 + t * 16 + nn;
      float bv = bias[n];
#pragma unroll
      for (int r = 0; r < 8; ++r) {
        int m = m0 + mi * 16 + r + 8 * hi;
        float v = acc[mi][t][r] + bv;
        if (RELU) v = fmaxf(v, 0.f);
        if (RES)  v += res[(size_t)m * N + n];
        if (OUTBF16) ((u16*)Cout)[(size_t)m * N + n] = f2bf(v);
        else         ((float*)Cout)[(size_t)m * N + n] = v;
      }
    }
  }
}

// ---------- flash attention with segment-causal mask, V = raw xcat ----------
// One wave per 16-row tile. qk: (L x 1024) bf16, q at cols 0..511, k at 512..1023.
__global__ __launch_bounds__(32) void attn_kernel(const u16* __restrict__ qk, const u16* __restrict__ xv,
                                                  const int* __restrict__ sep,
                                                  const int* __restrict__ seg_start,
                                                  float* __restrict__ ctx) {
  __shared__ alignas(16) float Os[16 * OSTR];
  __shared__ alignas(16) u16 Ps[16 * 32];
  __shared__ float lbuf[16];

  int lane = threadIdx.x;
  int nn = lane & 15, hi = lane >> 4;
  int i0 = blockIdx.x << 4;

  int srow[8];
  float mrow[8], lrow[8];
#pragma unroll
  for (int r = 0; r < 8; ++r) {
    srow[r] = sep[i0 + r + 8 * hi];
    mrow[r] = -1e30f;
    lrow[r] = 0.f;
  }
  for (int idx = lane; idx < 16 * OSTR; idx += 32) Os[idx] = 0.f;

  int imax = i0 + 15;
  int jstart = seg_start[sep[i0]] & ~31;
  const u16* kbase = qk + DMODEL;  // k columns start at 512

  for (int j0 = jstart; j0 <= imax; j0 += 32) {
    float sv[2][8];
    int   ok[2][8];
    // --- S = q . k^T for two 16-wide subtiles (16 wmma each over D=512) ---
#pragma unroll
    for (int c = 0; c < 2; ++c) {
      int jc = j0 + 16 * c;
      ffrag s;
#pragma unroll
      for (int r = 0; r < 8; ++r) s[r] = 0.f;
      for (int d0 = 0; d0 < DMODEL; d0 += 32) {
        bfrag qf = load_afrag(qk, i0, 1024, d0, lane);
        bfrag kf = load_bfrag(kbase, jc, 1024, d0, lane);
        s = wmma_bf16(qf, kf, s);
      }
      int j = jc + nn;          // this lane's column index (fixed for all 8 rows)
      int sj = sep[j];
#pragma unroll
      for (int r = 0; r < 8; ++r) {
        int i = i0 + r + 8 * hi;
        int good = (j <= i) && (sj == srow[r]);
        ok[c][r] = good;
        sv[c][r] = good ? s[r] : -1e30f;
      }
    }
    // --- online softmax: per-row max/sum via wave32 butterfly over 16 lanes ---
    float alpha[8], pv[2][8];
#pragma unroll
    for (int r = 0; r < 8; ++r) {
      float mx = fmaxf(sv[0][r], sv[1][r]);
      for (int d = 1; d < 16; d <<= 1) mx = fmaxf(mx, __shfl_xor(mx, d, 32));
      float mnew = fmaxf(mrow[r], mx);
      float p0 = ok[0][r] ? __expf(sv[0][r] - mnew) : 0.f;
      float p1 = ok[1][r] ? __expf(sv[1][r] - mnew) : 0.f;
      pv[0][r] = p0; pv[1][r] = p1;
      float sm = p0 + p1;
      for (int d = 1; d < 16; d <<= 1) sm += __shfl_xor(sm, d, 32);
      alpha[r] = __expf(mrow[r] - mnew);
      lrow[r] = lrow[r] * alpha[r] + sm;
      mrow[r] = mnew;
    }
    // --- stage P-tilde (16x32) into LDS in A-fragment order ---
#pragma unroll
    for (int c = 0; c < 2; ++c)
#pragma unroll
      for (int r = 0; r < 8; ++r)
        Ps[(r + 8 * hi) * 32 + c * 16 + nn] = f2bf(pv[c][r]);
    __asm volatile("s_wait_dscnt 0x0" ::: "memory");  // per-wave LDS ordering guard
    bfrag pf = load_pfrag(Ps, lane);
    // --- O = O*alpha + P~ @ V over all 32 d-tiles, O lives in LDS ---
    for (int nt = 0; nt < 32; ++nt) {
      bfrag vf = load_vfrag(xv, j0, nt * 16, lane);
      ffrag o;
#pragma unroll
      for (int r = 0; r < 8; ++r)
        o[r] = Os[(r + 8 * hi) * OSTR + nt * 16 + nn] * alpha[r];
      o = wmma_bf16(pf, vf, o);
#pragma unroll
      for (int r = 0; r < 8; ++r)
        Os[(r + 8 * hi) * OSTR + nt * 16 + nn] = o[r];
    }
  }
  // --- finalize: ctx = O / l ---
  if (nn == 0) {
#pragma unroll
    for (int r = 0; r < 8; ++r) lbuf[r + 8 * hi] = lrow[r];
  }
  __asm volatile("s_wait_dscnt 0x0" ::: "memory");
  for (int m = 0; m < 16; ++m) {
    float inv = 1.f / (lbuf[m] + 1e-10f);
    for (int d0 = lane; d0 < DMODEL; d0 += 32)
      ctx[(size_t)(i0 + m) * DMODEL + d0] = Os[m * OSTR + d0] * inv;
  }
}

// ---------- host launcher ----------
extern "C" void kernel_launch(void* const* d_in, const int* in_sizes, int n_in,
                              void* d_out, int out_size, void* d_ws, size_t ws_size,
                              hipStream_t stream) {
  (void)in_sizes; (void)n_in; (void)ws_size;
  const float* xcat = (const float*)d_in[0];
  const int*   xsep = (const int*)d_in[1];
  const float* Wqkv = (const float*)d_in[2];
  const float* bqkv = (const float*)d_in[3];
  const float* W1   = (const float*)d_in[4];
  const float* b1   = (const float*)d_in[5];
  const float* W2   = (const float*)d_in[6];
  const float* b2   = (const float*)d_in[7];
  float* out = (float*)d_out;

  // workspace carve-up (256B aligned)
  char* base = (char*)d_ws;
  size_t off = 0;
  auto carve = [&](size_t bytes) {
    char* p = base + off;
    off = (off + bytes + 255) & ~(size_t)255;
    return p;
  };
  float* mu1   = (float*)carve(DMODEL * 4);
  float* rs1   = (float*)carve(DMODEL * 4);
  float* mu2   = (float*)carve(DMODEL * 4);
  float* rs2   = (float*)carve(DMODEL * 4);
  int*   seg   = (int*)carve(16 * 4);
  u16*   xn    = (u16*)carve((size_t)LTOK * DMODEL * 2);       // LN(xcat) bf16
  u16*   xv    = (u16*)carve((size_t)LTOK * DMODEL * 2);       // xcat bf16 (attention V)
  u16*   qkb   = (u16*)carve((size_t)LTOK * 1024 * 2);         // [q | k] bf16
  float* ctx   = (float*)carve((size_t)LTOK * DMODEL * 4);
  u16*   ctxn  = (u16*)carve((size_t)LTOK * DMODEL * 2);       // LN(ctx) bf16
  u16*   hbuf  = (u16*)carve((size_t)LTOK * HFFN * 2);         // relu(ctxn@W1+b1) bf16
  u16*   Wqkt  = (u16*)carve((size_t)1024 * DMODEL * 2);       // Wqkv[:, :1024]^T bf16
  u16*   W1t   = (u16*)carve((size_t)HFFN * DMODEL * 2);       // W1^T bf16
  u16*   W2t   = (u16*)carve((size_t)DMODEL * HFFN * 2);       // W2^T bf16

  const int totLD = LTOK * DMODEL;

  // 1) column-LN stats + bf16 activations
  colstats_kernel<<<DMODEL, 256, 0, stream>>>(xcat, mu1, rs1, LTOK, DMODEL);
  ln_bf16_kernel<<<totLD / 256, 256, 0, stream>>>(xcat, mu1, rs1, xn, totLD);
  cvt_bf16_kernel<<<totLD / 256, 256, 0, stream>>>(xcat, xv, totLD);

  // 2) weight transpose+convert (one-time per launch, tiny)
  transpose_bf16_kernel<<<(1024 * DMODEL) / 256, 256, 0, stream>>>(Wqkv, 3 * DMODEL, 0, DMODEL, 1024, Wqkt);
  transpose_bf16_kernel<<<(HFFN * DMODEL) / 256, 256, 0, stream>>>(W1, HFFN, 0, DMODEL, HFFN, W1t);
  transpose_bf16_kernel<<<(DMODEL * HFFN) / 256, 256, 0, stream>>>(W2, DMODEL, 0, HFFN, DMODEL, W2t);

  // 3) qk = LN(x) @ Wqkv[:, :1024] + bqkv[:1024]  (v projection is dead in the reference)
  gemm_bf16<false, true, false><<<(LTOK / 32) * (1024 / 64) / 8, 256, 0, stream>>>(
      xn, Wqkt, bqkv, nullptr, qkb, LTOK, 1024, DMODEL);

  // 4) segment starts (xsep is sorted -> contiguous segments)
  seg_init_kernel<<<1, 16, 0, stream>>>(seg);
  seg_scan_kernel<<<LTOK / 256, 256, 0, stream>>>(xsep, seg, LTOK);

  // 5) flash attention with segment-causal mask; V = raw xcat
  attn_kernel<<<LTOK / 16, 32, 0, stream>>>(qkb, xv, xsep, seg, ctx);

  // 6) LN over ctx columns
  colstats_kernel<<<DMODEL, 256, 0, stream>>>(ctx, mu2, rs2, LTOK, DMODEL);
  ln_bf16_kernel<<<totLD / 256, 256, 0, stream>>>(ctx, mu2, rs2, ctxn, totLD);

  // 7) FFN: h = relu(ctxn @ W1 + b1);  out = h @ W2 + b2 + xcat
  gemm_bf16<true, true, false><<<(LTOK / 32) * (HFFN / 64) / 8, 256, 0, stream>>>(
      ctxn, W1t, b1, nullptr, hbuf, LTOK, HFFN, DMODEL);
  gemm_bf16<false, false, true><<<(LTOK / 32) * (DMODEL / 64) / 8, 256, 0, stream>>>(
      hbuf, W2t, b2, xcat, out, LTOK, DMODEL, HFFN);

  // 8) pass-through of xsep into the tail of d_out (raw bytes)
  if (out_size > totLD) {
    hipMemcpyAsync(out + totLD, d_in[1], (size_t)(out_size - totLD) * sizeof(float),
                   hipMemcpyDeviceToDevice, stream);
  }
}
